// VPNSFnet_13142599925818
// MI455X (gfx1250) — matrix operand order repaired and checked by reading, not compiled, not run on hardware
//
#include <hip/hip_runtime.h>
#include <math.h>

// ---------------------------------------------------------------------------
// Fused PINN residual for MI455X (gfx1250).
// 8 forward "streams" (primal, 4 first tangents, 3 second tangents) propagated
// through 7 hidden 256x256 tanh layers with v_wmma_f32_16x16x32_bf16.
// ~0.48 TFLOP vs ~3MB essential HBM traffic -> pure matrix compute.
// B fragments loaded via inline-asm ds_load_b128 (one 16-load clause) with
// staggered s_wait_dscnt chained through "+v" ties: the ties force 8 distinct
// live register tuples, giving the bulk clause + exact backend wait gating and
// no WMMA->VALU hazard nops. M=32 per workgroup so every B fragment feeds
// 2 WMMAs; weight panels streamed with CDNA5 async-to-LDS (ASYNCcnt).
// ---------------------------------------------------------------------------

#define NPTS    65536
#define HIDW    256
#define TM      32           // points per workgroup (2 WMMA m-tiles)
#define NHIDL   7            // hidden 256x256 WMMA layers (NHID-1)
#define NPANEL  (NHIDL * 16) // 112 weight column-panels of 16 cols each

typedef __bf16        v16bf __attribute__((ext_vector_type(16)));
typedef float         v8f   __attribute__((ext_vector_type(8)));
typedef unsigned int  v4u   __attribute__((ext_vector_type(4)));

struct Q16 { unsigned long long a, b; };     // 16-byte POD for b128 moves
union Frag  { Q16 q[2]; v16bf v; };          // A fragment (compiler loads)
union FragB { v4u d[2]; v16bf v; };          // B fragment (asm-pinned loads)

__device__ __forceinline__ unsigned short f2bf(float f) {
  union { float f; unsigned u; } x; x.f = f;
  unsigned r = x.u + 0x7FFFu + ((x.u >> 16) & 1u);  // round-to-nearest-even
  return (unsigned short)(r >> 16);
}
__device__ __forceinline__ float bf2f(unsigned short h) {
  union { unsigned u; float f; } x; x.u = ((unsigned)h) << 16;
  return x.f;
}

// CDNA5 async global->LDS copy (VGLOBAL op 98), tracked by ASYNCcnt.
__device__ __forceinline__ void async_ld16(unsigned ldsOff, unsigned long long ga) {
  asm volatile("global_load_async_to_lds_b128 %0, %1, off"
               :: "v"(ldsOff), "v"(ga) : "memory");
}
__device__ __forceinline__ void wait_async_le2() {
  asm volatile("s_wait_asynccnt 2" ::: "memory");
}
__device__ __forceinline__ void wait_async_0() {
  asm volatile("s_wait_asynccnt 0" ::: "memory");
}

// Convert hidden weights f32 [7][256][256] (k-major rows) to bf16 column-major
// Wb[l][c*256+k] so each WMMA B fragment is two contiguous 8-half LDS reads.
__global__ __launch_bounds__(256) void prep_weights(
    const float* __restrict__ Whid, unsigned short* __restrict__ Wb) {
  unsigned idx = blockIdx.x * 256u + threadIdx.x;     // dest index
  if (idx >= (unsigned)NHIDL * HIDW * HIDW) return;
  unsigned l = idx >> 16;
  unsigned rem = idx & 0xFFFFu;
  unsigned c = rem >> 8;
  unsigned k = rem & 0xFFu;
  Wb[idx] = f2bf(Whid[(size_t)l * HIDW * HIDW + k * HIDW + c]);
}

// LDS layout (dynamic, 290816 bytes; CDNA5 allows up to 320KB/workgroup):
//   act     : [2][8][32][256] bf16   @ 0       (262144 B, double buffered)
//   pbuf0   : [16][256]       bf16   @ 262144  (  8192 B, panel ping)
//   pbuf1   : [16][256]       bf16   @ 270336  (  8192 B, panel pong)
//   stage_d : [32][16]        f32    @ 278528  (  2048 B)
//   stage_zt: [3][32][16]     f32    @ 280576  (  6144 B)
//   outStage: [8][32][4]      f32    @ 286720  (  4096 B)
#define ACT_OFF    0
#define PBUF0_OFF  262144
#define PBUF1_OFF  270336
#define STD_OFF    278528
#define STZT_OFF   280576
#define OUTS_OFF   286720
#define SMEM_BYTES 290816

__global__ __launch_bounds__(256) void pinn_fused(
    const float* __restrict__ x,  const float* __restrict__ y,
    const float* __restrict__ z,  const float* __restrict__ t,
    const float* __restrict__ Win, const float* __restrict__ bin,
    const float* __restrict__ bhid, const unsigned short* __restrict__ Wb,
    const float* __restrict__ Wout, const float* __restrict__ bout,
    const float* __restrict__ lowb, const float* __restrict__ upb,
    float* __restrict__ out)
{
  extern __shared__ __align__(16) char smem[];
  unsigned short* act      = (unsigned short*)(smem + ACT_OFF);
  float*          stage_d  = (float*)(smem + STD_OFF);
  float*          stage_zt = (float*)(smem + STZT_OFF);
  float*          outStage = (float*)(smem + OUTS_OFF);

  const int tid  = threadIdx.x;
  const int lane = tid & 31;
  const int wave = tid >> 5;          // stream: 0=H 1..4=T_{x,y,z,t} 5..7=S_{x,y,z}
  const int hi   = (lane >> 4) & 1;   // WMMA half-wave split
  const int colL = lane & 15;         // N index within 16x16 tile
  const int p0   = blockIdx.x * TM;   // first point of this tile

  // LDS base addresses for asm ops: low 32 bits of the generic pointer are the
  // LDS offset (flat-aperture rule), keeping asm and C accesses consistent.
  const unsigned pbAddr[2] = { (unsigned)(size_t)(smem + PBUF0_OFF),
                               (unsigned)(size_t)(smem + PBUF1_OFF) };

  // Per-thread async panel copy: 32B of the 8KB panel (2 x b128 per thread).
  const unsigned pbOffT[2] = { pbAddr[0] + (unsigned)tid * 32,
                               pbAddr[1] + (unsigned)tid * 32 };
  const unsigned long long WbT = (unsigned long long)(const char*)Wb + tid * 32;
  #define ISSUE_PANEL(g) do {                                              \
      unsigned lo = pbOffT[(g) & 1];                                       \
      unsigned long long ga = WbT + ((unsigned long long)(g) << 13);       \
      async_ld16(lo, ga);                                                  \
      async_ld16(lo + 16, ga + 16);                                        \
    } while (0)

  // Prologue: start panels 0 and 1 streaming while we do the input layer.
  ISSUE_PANEL(0);
  ISSUE_PANEL(1);

  // -------- input layer: [4]->[256], scalar (K=4, not worth WMMA) --------
  {
    int r = tid >> 3;                 // 8 threads per point row, 32 rows
    int p = p0 + r;
    float Xv[4] = { x[p], y[p], z[p], t[p] };
    float g[4], xn[4];
    #pragma unroll
    for (int j = 0; j < 4; ++j) {
      g[j]  = 2.0f / (upb[j] - lowb[j]);
      xn[j] = (Xv[j] - lowb[j]) * g[j] - 1.0f;
    }
    int c0 = (tid & 7) << 5;
    for (int i = 0; i < 32; ++i) {
      int c = c0 + i;
      float zp = bin[c];
      float gw[4];
      #pragma unroll
      for (int j = 0; j < 4; ++j) {
        float wv = Win[j * HIDW + c];
        zp += xn[j] * wv;
        gw[j] = g[j] * wv;            // dz/dx_j along unit direction
      }
      float a = tanhf(zp);
      float d = 1.0f - a * a;
      act[(0 * 8 + 0) * 8192 + r * 256 + c] = f2bf(a);
      #pragma unroll
      for (int j = 0; j < 4; ++j)
        act[(0 * 8 + 1 + j) * 8192 + r * 256 + c] = f2bf(d * gw[j]);
      #pragma unroll
      for (int j = 0; j < 3; ++j)
        act[(0 * 8 + 5 + j) * 8192 + r * 256 + c] =
            f2bf(-2.0f * a * d * gw[j] * gw[j]);
    }
  }
  wait_async_le2();                   // panel 0 resident (in-order completion)
  __syncthreads();                    // act[0] + panel 0 visible to all waves

  // -------- 7 hidden layers: flat pipeline over 112 column panels --------
  Frag Afr[2][8];                     // hoisted A fragments, 2 m-tiles x K=256
  for (int g = 0; g < NPANEL; ++g) {
    const int l   = g >> 4;
    const int n   = g & 15;
    const int pb  = g & 1;
    const int cur = l & 1;
    const int nxt = cur ^ 1;

    if (n == 0) {                     // (re)load A fragments once per layer
      #pragma unroll
      for (int m = 0; m < 2; ++m) {
        const unsigned short* ap =
            act + (cur * 8 + wave) * 8192 + (m * 16 + colL) * 256;
        #pragma unroll
        for (int kt = 0; kt < 8; ++kt) {
          int abase = kt * 32 + hi * 8;
          Afr[m][kt].q[0] = *(const Q16*)(ap + abase);
          Afr[m][kt].q[1] = *(const Q16*)(ap + abase + 16);
        }
      }
    }

    // Asm-pinned bulk load of all 8 B fragments: one 16-wide ds clause.
    FragB Bfr[8];
    const unsigned baddr = pbAddr[pb] + (unsigned)(colL * 256 + hi * 8) * 2;
    #pragma unroll
    for (int kt = 0; kt < 8; ++kt) {
      asm volatile("ds_load_b128 %0, %2\n\t"
                   "ds_load_b128 %1, %2 offset:32"
                   : "=v"(Bfr[kt].d[0]), "=v"(Bfr[kt].d[1])
                   : "v"(baddr + (unsigned)kt * 64));
    }

    v8f acc[2];
    acc[0] = (v8f){0.f, 0.f, 0.f, 0.f, 0.f, 0.f, 0.f, 0.f};
    acc[1] = (v8f){0.f, 0.f, 0.f, 0.f, 0.f, 0.f, 0.f, 0.f};
    #pragma unroll
    for (int kt = 0; kt < 8; ++kt) {  // K = 256 in 8 x 32 steps
      // Staggered wait chained through the fragment registers: keeps all 8
      // B tuples distinct/live (bulk clause) and gates the WMMA pair.
      asm volatile("s_wait_dscnt %2"
                   : "+v"(Bfr[kt].d[0]), "+v"(Bfr[kt].d[1])
                   : "n"(14 - 2 * kt));
      acc[0] = __builtin_amdgcn_wmma_f32_16x16x32_bf16(
                   false, Afr[0][kt].v, false, Bfr[kt].v,
                   (short)0, acc[0], false, false);
      acc[1] = __builtin_amdgcn_wmma_f32_16x16x32_bf16(
                   false, Afr[1][kt].v, false, Bfr[kt].v,
                   (short)0, acc[1], false, false);
    }

    // -------- elementwise derivative combine --------
    // C layout: VGPR r, lane -> element (row = m*16 + r + hi*8, col = colL)
    const int colG = n * 16 + colL;
    if (wave == 0) {                  // primal: a, d
      float bb = bhid[l * HIDW + colG];
      #pragma unroll
      for (int m = 0; m < 2; ++m) {
        #pragma unroll
        for (int r = 0; r < 8; ++r) {
          int row = m * 16 + r + hi * 8;
          float a = tanhf(acc[m][r] + bb);
          float d = 1.0f - a * a;
          stage_d[row * 16 + colL] = d;
          act[(nxt * 8 + 0) * 8192 + row * 256 + colG] = f2bf(a);
        }
      }
    } else if (wave <= 3) {           // spatial tangents: stage zT_i
      #pragma unroll
      for (int m = 0; m < 2; ++m) {
        #pragma unroll
        for (int r = 0; r < 8; ++r) {
          int row = m * 16 + r + hi * 8;
          stage_zt[(wave - 1) * 512 + row * 16 + colL] = acc[m][r];
        }
      }
    }
    __syncthreads();
    if (wave >= 1 && wave <= 4) {     // T_j' = d * zT_j
      #pragma unroll
      for (int m = 0; m < 2; ++m) {
        #pragma unroll
        for (int r = 0; r < 8; ++r) {
          int row = m * 16 + r + hi * 8;
          float d = stage_d[row * 16 + colL];
          act[(nxt * 8 + wave) * 8192 + row * 256 + colG] = f2bf(d * acc[m][r]);
        }
      }
    } else if (wave >= 5) {           // S_i' = d*zS_i - 2 a d zT_i^2
      const int i = wave - 5;
      #pragma unroll
      for (int m = 0; m < 2; ++m) {
        #pragma unroll
        for (int r = 0; r < 8; ++r) {
          int row = m * 16 + r + hi * 8;
          float d  = stage_d[row * 16 + colL];
          float a  = bf2f(act[(nxt * 8 + 0) * 8192 + row * 256 + colG]);
          float zt = stage_zt[i * 512 + row * 16 + colL];
          act[(nxt * 8 + wave) * 8192 + row * 256 + colG] =
              f2bf(d * acc[m][r] - 2.0f * a * d * zt * zt);
        }
      }
    }

    __syncthreads();                  // all reads of pbuf[pb]/stage complete
    if (g < NPANEL - 2) {             // stream panel g+2 into the freed buffer
      ISSUE_PANEL(g + 2);
      wait_async_le2();               // panel g+1 resident (only g+2 in flight)
    } else {
      wait_async_0();                 // drain tail
    }
    __syncthreads();                  // next panel visible to all waves
  }

  // -------- output layer: [256]->[4], scalar dots (N=4) --------
  const int fin = NHIDL & 1;          // final activation buffer
  {
    float* WoutL = (float*)(smem + PBUF0_OFF);   // reuse 8KB panel region
    #pragma unroll
    for (int i = 0; i < 4; ++i) WoutL[tid + 256 * i] = Wout[tid + 256 * i];
    __syncthreads();
    #pragma unroll
    for (int qq = 0; qq < 4; ++qq) {  // 1024 dot products / 256 threads
      int q = tid * 4 + qq;
      int s = q >> 7, r = (q >> 2) & 31, c = q & 3;
      const unsigned short* ap = act + (fin * 8 + s) * 8192 + r * 256;
      float sum = (s == 0) ? bout[c] : 0.0f;
      for (int k = 0; k < 256; ++k) sum += bf2f(ap[k]) * WoutL[k * 4 + c];
      outStage[q] = sum;
    }
  }
  __syncthreads();

  // -------- Navier-Stokes residual combine, write 8 outputs --------
  if (tid < TM) {
    int r = tid, p = p0 + r;
    #define O(s, c) outStage[((s) << 7) | (r << 2) | (c)]
    float u  = O(0,0), v  = O(0,1), w  = O(0,2), pr = O(0,3);
    float ux = O(1,0), vx = O(1,1), wx = O(1,2), px = O(1,3);
    float uy = O(2,0), vy = O(2,1), wy = O(2,2), py = O(2,3);
    float uz = O(3,0), vz = O(3,1), wz = O(3,2), pz = O(3,3);
    float ut = O(4,0), vt = O(4,1), wt = O(4,2);
    float lapu = O(5,0) + O(6,0) + O(7,0);
    float lapv = O(5,1) + O(6,1) + O(7,1);
    float lapw = O(5,2) + O(6,2) + O(7,2);
    #undef O
    out[0 * NPTS + p] = u;
    out[1 * NPTS + p] = v;
    out[2 * NPTS + p] = w;
    out[3 * NPTS + p] = pr;
    out[4 * NPTS + p] = ut + (u * ux + v * uy + w * uz) + px - lapu;
    out[5 * NPTS + p] = vt + (u * vx + v * vy + w * vz) + py - lapv;
    out[6 * NPTS + p] = wt + (u * wx + v * wy + w * wz) + pz - lapw;
    out[7 * NPTS + p] = ux + vy + wz;
  }
}

extern "C" void kernel_launch(void* const* d_in, const int* in_sizes, int n_in,
                              void* d_out, int out_size, void* d_ws, size_t ws_size,
                              hipStream_t stream) {
  (void)in_sizes; (void)n_in; (void)out_size; (void)ws_size;
  const float* x    = (const float*)d_in[0];
  const float* y    = (const float*)d_in[1];
  const float* z    = (const float*)d_in[2];
  const float* t    = (const float*)d_in[3];
  const float* Win  = (const float*)d_in[4];
  const float* bin  = (const float*)d_in[5];
  const float* Whid = (const float*)d_in[6];
  const float* bhid = (const float*)d_in[7];
  const float* Wout = (const float*)d_in[8];
  const float* bout = (const float*)d_in[9];
  const float* lowb = (const float*)d_in[10];
  const float* upb  = (const float*)d_in[11];

  unsigned short* Wb = (unsigned short*)d_ws;     // 7*256*256 bf16 = 896 KB

  prep_weights<<<(NHIDL * HIDW * HIDW) / 256, 256, 0, stream>>>(Whid, Wb);
  pinn_fused<<<NPTS / TM, 256, SMEM_BYTES, stream>>>(
      x, y, z, t, Win, bin, bhid, Wb, Wout, bout, lowb, upb, (float*)d_out);
}
